// DecoderLSTM_50251117363499
// MI455X (gfx1250) — matrix operand order repaired and checked by reading, not compiled, not run on hardware
//
#include <hip/hip_runtime.h>
#include <hip/hip_bf16.h>

#define BATCH   4096
#define HIDN    1024
#define EMBN    512
#define VOCABN  23
#define LATENTN 128
#define CONDN   16
#define INPN    144   // LATENT + COND
#define MAXLEN  64
#define PADIDX  21
#define SOSIDX  22
#define BLOCK_M 32

typedef __bf16 bf16;
typedef __bf16 v16bf __attribute__((ext_vector_type(16)));
typedef __bf16 v8bf  __attribute__((ext_vector_type(8)));
typedef float  v8f   __attribute__((ext_vector_type(8)));

// A-fragment (16x32 bf16, MxK) per CDNA5 layout:
//   lanes 0-15 : elements 0..7 = K 0..7,   elements 8..15 = K 16..23
//   lanes16-31 : elements 0..7 = K 8..15,  elements 8..15 = K 24..31
// caller passes p = &row[kb + hi*8]; second half at +16 elements.
__device__ __forceinline__ v16bf lds_a_frag(const bf16* p) {
  v8bf lo = *(const v8bf*)p;
  v8bf hh = *(const v8bf*)(p + 16);
  v16bf r;
#pragma unroll
  for (int i = 0; i < 8; ++i) { r[i] = lo[i]; r[i + 8] = hh[i]; }
  return r;
}

__device__ __forceinline__ float fsigmoid(float x) { return 1.f / (1.f + __expf(-x)); }
__device__ __forceinline__ float ftanh(float x)    { return 1.f - 2.f / (__expf(2.f * x) + 1.f); }

#define WMMA_BF16(A, Bf, C) \
  __builtin_amdgcn_wmma_f32_16x16x32_bf16(false, (A), false, (Bf), (short)0, (C), false, false)

// ---------------------------------------------------------------------------
// Prep: fp32 -> bf16 weight conversion, W_out padded to 32 rows, fused bias.
// ---------------------------------------------------------------------------
__global__ void DecoderLSTM_prep(const float* __restrict__ Wih,
                                 const float* __restrict__ Whh,
                                 const float* __restrict__ Wout,
                                 const float* __restrict__ bih,
                                 const float* __restrict__ bhh,
                                 bf16* __restrict__ WihB, bf16* __restrict__ WhhB,
                                 bf16* __restrict__ WoutB, float* __restrict__ biasWS) {
  int i = blockIdx.x * 256 + threadIdx.x;
  if (i < 4 * HIDN * EMBN) WihB[i] = (bf16)Wih[i];
  if (i < 4 * HIDN * HIDN) WhhB[i] = (bf16)Whh[i];
  if (i < 32 * HIDN) {
    int r = i / HIDN;
    WoutB[i] = (r < VOCABN) ? (bf16)Wout[i] : (bf16)0.f;
  }
  if (i < 4 * HIDN) biasWS[i] = bih[i] + bhh[i];
}

// ---------------------------------------------------------------------------
// Persistent decoder: one block owns 32 batch rows for all 64 steps.
// ---------------------------------------------------------------------------
__global__ __launch_bounds__(256, 1)
void DecoderLSTM_main(const float* __restrict__ z, const float* __restrict__ cnd,
                      const float* __restrict__ Etab,
                      const float* __restrict__ Wlh, const float* __restrict__ blh,
                      const bf16* __restrict__ WihB, const bf16* __restrict__ WhhB,
                      const float* __restrict__ biasWS,
                      const bf16* __restrict__ WoutB, const float* __restrict__ bout,
                      float* __restrict__ out) {
  __shared__ __align__(16) bf16  sX[BLOCK_M][EMBN];        // 32 KB, current token embeddings
  __shared__ __align__(16) bf16  sH[2][BLOCK_M][HIDN];     // 128 KB, double-buffered hidden
  __shared__ __align__(16) float sC[BLOCK_M][HIDN];        // 128 KB, cell state
  __shared__ __align__(16) float sLog[BLOCK_M][32];        // 4 KB, logits (padded to 32)
  __shared__ int sTok[BLOCK_M];

  const int tid   = threadIdx.x;
  const int lane  = tid & 31;
  const int wave  = tid >> 5;
  const int m16   = lane & 15;   // N-col (B/C) or M-row (A) within tile
  const int hi    = lane >> 4;
  const int mbase = blockIdx.x * BLOCK_M;

  // ---- stage [z | cond] fp32 into alias of sX (18KB <= 32KB) ----
  float* sIn = (float*)&sX[0][0];
  for (int idx = tid; idx < BLOCK_M * INPN; idx += 256) {
    int m = idx / INPN, k = idx - m * INPN;
    sIn[idx] = (k < LATENTN) ? z[(mbase + m) * LATENTN + k]
                             : cnd[(mbase + m) * CONDN + (k - LATENTN)];
  }
  __syncthreads();

  // ---- h0 = [z|cond] @ W_lh^T + b_lh  (one-time, plain FMA) ----
  for (int j = tid; j < HIDN; j += 256) {
    const float* wrow = &Wlh[j * INPN];
    float bj = blh[j];
    for (int m = 0; m < BLOCK_M; ++m) {
      const float* xr = &sIn[m * INPN];
      float acc = bj;
      for (int k = 0; k < INPN; ++k) acc = fmaf(xr[k], wrow[k], acc);
      sH[0][m][j] = (bf16)acc;
    }
  }
  __syncthreads();

  // ---- c0 = 0, x0 = E[SOS] (overwrites sIn alias region, post-barrier) ----
  for (int idx = tid; idx < BLOCK_M * HIDN; idx += 256)
    sC[idx >> 10][idx & (HIDN - 1)] = 0.f;
  for (int idx = tid; idx < BLOCK_M * EMBN; idx += 256) {
    int m = idx >> 9, k = idx & (EMBN - 1);
    sX[m][k] = (bf16)Etab[SOSIDX * EMBN + k];
  }
  __syncthreads();

  for (int t = 0; t < MAXLEN; ++t) {
    const int rd = t & 1, wr = (t + 1) & 1;

    // ===== Phase A: logits init + gate GEMM (WMMA bf16) + LSTM cell =====
    for (int idx = tid; idx < BLOCK_M * 32; idx += 256) {
      int m = idx >> 5, v = idx & 31;
      sLog[m][v] = (v < VOCABN) ? bout[v] : 0.f;
    }

    for (int chunk = 0; chunk < 8; ++chunk) {
      const int jb   = wave * 128 + chunk * 16;  // hidden-unit tile base
      const int jcol = jb + m16;
      v8f acc[2][4];
#pragma unroll
      for (int q = 0; q < 4; ++q) {
        float bq = biasWS[q * HIDN + jcol];
#pragma unroll
        for (int mt = 0; mt < 2; ++mt)
#pragma unroll
          for (int r = 0; r < 8; ++r) acc[mt][q][r] = bq;
      }
      // K over EMB: x @ W_ih^T
      for (int kb = 0; kb < EMBN; kb += 32) {
        v16bf a0 = lds_a_frag(&sX[m16][kb + hi * 8]);
        v16bf a1 = lds_a_frag(&sX[16 + m16][kb + hi * 8]);
#pragma unroll
        for (int q = 0; q < 4; ++q) {
          v16bf bf = lds_a_frag(&WihB[(q * HIDN + jb + m16) * EMBN + kb + hi * 8]);
          acc[0][q] = WMMA_BF16(a0, bf, acc[0][q]);
          acc[1][q] = WMMA_BF16(a1, bf, acc[1][q]);
        }
      }
      // K over HID: h @ W_hh^T
      for (int kb = 0; kb < HIDN; kb += 32) {
        v16bf a0 = lds_a_frag(&sH[rd][m16][kb + hi * 8]);
        v16bf a1 = lds_a_frag(&sH[rd][16 + m16][kb + hi * 8]);
#pragma unroll
        for (int q = 0; q < 4; ++q) {
          v16bf bf = lds_a_frag(&WhhB[(q * HIDN + jb + m16) * HIDN + kb + hi * 8]);
          acc[0][q] = WMMA_BF16(a0, bf, acc[0][q]);
          acc[1][q] = WMMA_BF16(a1, bf, acc[1][q]);
        }
      }
      // LSTM elementwise: C-tile layout -> lane holds M = mt*16 + r + hi*8, N = jcol
#pragma unroll
      for (int mt = 0; mt < 2; ++mt)
#pragma unroll
        for (int r = 0; r < 8; ++r) {
          int m = mt * 16 + r + hi * 8;
          float iv = fsigmoid(acc[mt][0][r]);
          float fv = fsigmoid(acc[mt][1][r]);
          float gv = ftanh(acc[mt][2][r]);
          float ov = fsigmoid(acc[mt][3][r]);
          float cv = fv * sC[m][jcol] + iv * gv;
          sC[m][jcol] = cv;
          sH[wr][m][jcol] = (bf16)(ov * ftanh(cv));
        }
    }
    __syncthreads();

    // ===== Phase B: logits = h_new @ W_out^T via WMMA, K split over waves =====
    {
      v8f lacc[2][2];
#pragma unroll
      for (int mt = 0; mt < 2; ++mt)
#pragma unroll
        for (int nt = 0; nt < 2; ++nt)
#pragma unroll
          for (int r = 0; r < 8; ++r) lacc[mt][nt][r] = 0.f;
#pragma unroll
      for (int kk = 0; kk < 4; ++kk) {
        int kb = wave * 128 + kk * 32;
        v16bf a0 = lds_a_frag(&sH[wr][m16][kb + hi * 8]);
        v16bf a1 = lds_a_frag(&sH[wr][16 + m16][kb + hi * 8]);
#pragma unroll
        for (int nt = 0; nt < 2; ++nt) {
          v16bf bf = lds_a_frag(&WoutB[(nt * 16 + m16) * HIDN + kb + hi * 8]);
          lacc[0][nt] = WMMA_BF16(a0, bf, lacc[0][nt]);
          lacc[1][nt] = WMMA_BF16(a1, bf, lacc[1][nt]);
        }
      }
#pragma unroll
      for (int mt = 0; mt < 2; ++mt)
#pragma unroll
        for (int nt = 0; nt < 2; ++nt)
#pragma unroll
          for (int r = 0; r < 8; ++r)
            atomicAdd(&sLog[mt * 16 + r + hi * 8][nt * 16 + m16], lacc[mt][nt][r]);
    }
    __syncthreads();

    // ===== Phase C: greedy argmax (first-max) + store logits =====
    if (tid < BLOCK_M) {
      int best = 0;
      float bv = sLog[tid][0];
#pragma unroll
      for (int v = 1; v < VOCABN; ++v) {
        float x = sLog[tid][v];
        if (x > bv) { bv = x; best = v; }
      }
      sTok[tid] = best;
    } else {
      for (int id = tid - 32; id < BLOCK_M * VOCABN; id += 224) {
        int m = id / VOCABN, v = id - m * VOCABN;
        out[(mbase + m) * (MAXLEN * VOCABN) + t * VOCABN + v] = sLog[m][v];
      }
    }
    __syncthreads();

    // ===== Phase D: x = E[token]  (PAD row forced to zero) =====
    for (int idx = tid; idx < BLOCK_M * EMBN; idx += 256) {
      int m = idx >> 9, k = idx & (EMBN - 1);
      int tok = sTok[m];
      sX[m][k] = (bf16)((tok == PADIDX) ? 0.f : Etab[tok * EMBN + k]);
    }
    __syncthreads();
  }
}

// ---------------------------------------------------------------------------
extern "C" void kernel_launch(void* const* d_in, const int* in_sizes, int n_in,
                              void* d_out, int out_size, void* d_ws, size_t ws_size,
                              hipStream_t stream) {
  (void)in_sizes; (void)n_in; (void)out_size; (void)ws_size;
  const float* z    = (const float*)d_in[0];
  const float* cnd  = (const float*)d_in[1];
  const float* Etab = (const float*)d_in[2];
  const float* Wlh  = (const float*)d_in[3];
  const float* blh  = (const float*)d_in[4];
  const float* Wih  = (const float*)d_in[5];
  const float* Whh  = (const float*)d_in[6];
  const float* bih  = (const float*)d_in[7];
  const float* bhh  = (const float*)d_in[8];
  const float* Wout = (const float*)d_in[9];
  const float* bout = (const float*)d_in[10];
  float* out = (float*)d_out;

  // workspace layout: bf16 W_ih | bf16 W_hh | bf16 W_out(32 rows) | fused bias
  bf16*  WihB   = (bf16*)d_ws;
  bf16*  WhhB   = WihB + 4 * HIDN * EMBN;          // +4 MB
  bf16*  WoutB  = WhhB + 4 * HIDN * HIDN;          // +8 MB
  float* biasWS = (float*)(WoutB + 32 * HIDN);     // +64 KB  (total ~12.7 MB)

  const int nconv = 4 * HIDN * HIDN;               // largest conversion
  DecoderLSTM_prep<<<(nconv + 255) / 256, 256, 0, stream>>>(
      Wih, Whh, Wout, bih, bhh, WihB, WhhB, WoutB, biasWS);

  DecoderLSTM_main<<<BATCH / BLOCK_M, 256, 0, stream>>>(
      z, cnd, Etab, Wlh, blh, WihB, WhhB, biasWS, WoutB, bout, out);
}
